// CausalSelfAttentionWithKVCache_52424370815065
// MI455X (gfx1250) — compile-verified, matrix operand order checked by reading, not run on hardware
//
#include <hip/hip_runtime.h>
#include <hip/hip_bf16.h>

typedef __attribute__((ext_vector_type(8)))  __bf16 v8bf;
typedef __attribute__((ext_vector_type(16))) __bf16 v16bf;
typedef __attribute__((ext_vector_type(8)))  float  v8f;
typedef __attribute__((ext_vector_type(4))) unsigned int v4u;
typedef __attribute__((ext_vector_type(8))) int v8i_t;
typedef __attribute__((ext_vector_type(4))) int v4i_t;

#define NEG_BIG (-1e30f)
#define LDSROW 40          // padded LDS row stride in bf16 elems (80 B, conflict-free)
#define TILE_BYTES (128 * LDSROW * 2)  // one 128x32 panel with padding: 10240 B
#define DEPTH 3            // TDM pipeline depth (triple buffer)

#if __has_builtin(__builtin_amdgcn_tensor_load_to_lds)
#define HAVE_TDM 1
#else
#define HAVE_TDM 0
#endif

// D = A(16x32 bf16) * B(32x16 bf16) + C(16x16 f32), wave32 WMMA
__device__ __forceinline__ v8f wmma_bf16(v16bf a, v16bf b, v8f c) {
  return __builtin_amdgcn_wmma_f32_16x16x32_bf16(
      false, a, false, b, (short)0, c, false, false);
}

// 16x32 bf16 fragment from row-major global memory (A-style; B mirrors with
// "row" = N column). Lanes 0-15: row r, K {0..7,16..23}; lanes 16-31: row r,
// K {8..15,24..31} -> two contiguous 16-byte chunks.
__device__ __forceinline__ v16bf load_frag(const __bf16* base, int row,
                                           long long ld, int col, int lane) {
  const int r = lane & 15, half = lane >> 4;
  const __bf16* p = base + (long long)(row + r) * ld + col + half * 8;
  v8bf lo = *(const v8bf*)p;
  v8bf hi = *(const v8bf*)(p + 16);
  return __builtin_shufflevector(lo, hi, 0, 1, 2, 3, 4, 5, 6, 7,
                                 8, 9, 10, 11, 12, 13, 14, 15);
}

// Same fragment gather from a padded (LDSROW-stride) LDS panel.
__device__ __forceinline__ v16bf lds_frag(const __bf16* t, int row, int lane) {
  const int r = lane & 15, half = lane >> 4;
  const __bf16* p = t + (row + r) * LDSROW + half * 8;
  v8bf lo = *(const v8bf*)p;
  v8bf hi = *(const v8bf*)(p + 16);
  return __builtin_shufflevector(lo, hi, 0, 1, 2, 3, 4, 5, 6, 7,
                                 8, 9, 10, 11, 12, 13, 14, 15);
}

__global__ void cvt_kernel(const float* __restrict__ in,
                           __bf16* __restrict__ out, int n) {
  int i = blockIdx.x * blockDim.x + threadIdx.x;
  int stride = gridDim.x * blockDim.x;
  for (; i < n; i += stride) out[i] = (__bf16)in[i];
}

#if HAVE_TDM
// TDM: DMA one 128-row x 32-elem bf16 panel (row stride `ld` elems) into LDS
// at byte offset lds_off, inserting 16 B of pad per 64 B row -> LDSROW stride.
__device__ __forceinline__ void tdm_load_128x32(const __bf16* gsrc,
                                                unsigned lds_off, unsigned ld) {
  const unsigned long long ga = (unsigned long long)(const void*)gsrc;
  v4u g0;
  g0[0] = 1u;                                   // count=1, user mode
  g0[1] = lds_off;                              // lds_addr
  g0[2] = (unsigned)(ga & 0xffffffffu);         // global_addr[31:0]
  g0[3] = (unsigned)((ga >> 32) & 0x01ffffffu) | (2u << 30);  // addr hi | type=2
  const unsigned td0 = ld;            // tensor_dim0 (row length)
  const unsigned td1 = 0x00ffffffu;   // tensor_dim1 (rows; generous, no OOB)
  v8i_t g1;
  // data_size=1 (2B) | pad_enable | pad_interval=3 (16 DW) | pad_amount=3 (4 DW)
  g1[0] = (int)((1u << 16) | (1u << 20) | (3u << 22) | (3u << 25));
  g1[1] = (int)((td0 & 0xffffu) << 16);                        // dim0 lo
  g1[2] = (int)(((td0 >> 16) & 0xffffu) | ((td1 & 0xffffu) << 16));
  g1[3] = (int)(((td1 >> 16) & 0xffffu) | (32u << 16));        // tile_dim0=32
  g1[4] = 128;                                                 // tile_dim1=128
  g1[5] = (int)td0;                                            // dim0_stride lo
  g1[6] = 0;
  g1[7] = 0;
  v4i_t z4 = {0, 0, 0, 0};
#if __has_include(<hip/amd_detail/amd_gfx1250_TDM.h>)
  v8i_t z8 = {0, 0, 0, 0, 0, 0, 0, 0};
  __builtin_amdgcn_tensor_load_to_lds(g0, g1, z4, z4, z8, 0);
#else
  __builtin_amdgcn_tensor_load_to_lds(g0, g1, z4, z4, 0);
#endif
}
#else
// Fallback staging: cooperative 256-thread copy, 16 B chunks, padded stride.
__device__ __forceinline__ void stage_tile(__bf16* dst, const __bf16* src,
                                           long long ld) {
  for (int c = threadIdx.x; c < 512; c += 256) {
    const int row = c >> 2, q = c & 3;
    *(v8bf*)(dst + row * LDSROW + q * 8) =
        *(const v8bf*)(src + (long long)row * ld + q * 8);
  }
}
#endif

// Block-tiled GEMM mainloop: block = 128(M) x 128(N), K = 2048 in steps of 32.
// 8 waves arranged 4x2; each wave owns a 32x64 tile -> acc[2][4] (64 VGPRs).
__device__ __forceinline__ void gemm_mainloop(const __bf16* __restrict__ A,
                                              const __bf16* __restrict__ Bw,
                                              int row0, int col0, __bf16* shA,
                                              __bf16* shB, v8f acc[2][4]) {
  const int lane = threadIdx.x & 31;
  const int wv = threadIdx.x >> 5;
  const int wm = (wv >> 1) * 32;
  const int wn = (wv & 1) * 64;
  const int nsteps = 2048 / 32;  // 64
#if HAVE_TDM
  const unsigned offA = (unsigned)(unsigned long long)(const void*)shA;
  const unsigned offB = (unsigned)(unsigned long long)(const void*)shB;
  if (wv == 0) {
#pragma unroll
    for (int s = 0; s < DEPTH; ++s) {
      tdm_load_128x32(A + (long long)row0 * 2048 + s * 32, offA + s * TILE_BYTES, 2048);
      tdm_load_128x32(Bw + (long long)col0 * 2048 + s * 32, offB + s * TILE_BYTES, 2048);
    }
  }
#endif
  for (int k = 0; k < nsteps; ++k) {
#if HAVE_TDM
    if (wv == 0) {  // wait for the oldest (slot k) pair; TENSORcnt is in-order
      if (k + 1 >= nsteps)      __builtin_amdgcn_s_wait_tensorcnt(0);
      else if (k + 2 >= nsteps) __builtin_amdgcn_s_wait_tensorcnt(2);
      else                      __builtin_amdgcn_s_wait_tensorcnt(4);
    }
    __syncthreads();
    const int slot = k % DEPTH;
    const __bf16* a0 = shA + slot * (128 * LDSROW);
    const __bf16* b0 = shB + slot * (128 * LDSROW);
#else
    __syncthreads();
    stage_tile(shA, A + (long long)row0 * 2048 + k * 32, 2048);
    stage_tile(shB, Bw + (long long)col0 * 2048 + k * 32, 2048);
    __syncthreads();
    const __bf16* a0 = shA;
    const __bf16* b0 = shB;
#endif
    v16bf af[2], bfr[4];
#pragma unroll
    for (int i = 0; i < 2; ++i) af[i] = lds_frag(a0, wm + 16 * i, lane);
#pragma unroll
    for (int j = 0; j < 4; ++j) bfr[j] = lds_frag(b0, wn + 16 * j, lane);
#pragma unroll
    for (int i = 0; i < 2; ++i)
#pragma unroll
      for (int j = 0; j < 4; ++j)
        acc[i][j] = wmma_bf16(af[i], bfr[j], acc[i][j]);
#if HAVE_TDM
    __syncthreads();  // all waves done with slot k before its buffer is reused
    if (wv == 0 && (k + DEPTH) < nsteps) {
      const int ns = (k + DEPTH) % DEPTH;  // == slot just freed
      tdm_load_128x32(A + (long long)row0 * 2048 + (k + DEPTH) * 32,
                      offA + ns * TILE_BYTES, 2048);
      tdm_load_128x32(Bw + (long long)col0 * 2048 + (k + DEPTH) * 32,
                      offB + ns * TILE_BYTES, 2048);
    }
#endif
  }
}

// QKV GEMM: M=8192 (B*T), N=6144 (3C), K=2048.
__global__ void __launch_bounds__(256) qkv_kernel(
    const __bf16* __restrict__ xbf, const __bf16* __restrict__ wbf,
    const float* __restrict__ bias, __bf16* __restrict__ qf,
    __bf16* __restrict__ kf, __bf16* __restrict__ vT,
    float* __restrict__ kout, float* __restrict__ vout) {
  __shared__ __align__(16) __bf16 shA[DEPTH * 128 * LDSROW];
  __shared__ __align__(16) __bf16 shB[DEPTH * 128 * LDSROW];
  const int tm = blockIdx.x / 48, tn = blockIdx.x % 48;
  const int row0 = tm * 128, col0 = tn * 128;
  v8f acc[2][4] = {};
  gemm_mainloop(xbf, wbf, row0, col0, shA, shB, acc);

  const int lane = threadIdx.x & 31;
  const int wv = threadIdx.x >> 5;
  const int wm = (wv >> 1) * 32, wn = (wv & 1) * 64;
  const int n = lane & 15, half = lane >> 4;
  const float qscale = 0.08838834764831845f;  // 1/sqrt(128)
#pragma unroll
  for (int i = 0; i < 2; ++i) {
#pragma unroll
    for (int j = 0; j < 4; ++j) {
#pragma unroll
      for (int jj = 0; jj < 8; ++jj) {
        const int m = jj + 8 * half;
        const int grow = row0 + wm + 16 * i + m;  // flat b*T + t
        const int o = col0 + wn + 16 * j + n;     // [0, 6144)
        const float val = acc[i][j][jj] + bias[o];
        const int bb = grow >> 11, t = grow & 2047;
        if (o < 2048) {
          qf[(long long)grow * 2048 + o] = (__bf16)(val * qscale);
        } else if (o < 4096) {
          const int c = o - 2048, h = c >> 7, d = c & 127;
          kf[(long long)grow * 2048 + c] = (__bf16)val;
          kout[((long long)((bb * 16 + h) * 2048 + t)) * 128 + d] = val;
        } else {
          const int c = o - 4096, h = c >> 7, d = c & 127;
          vT[((long long)(bb * 16 + h) * 128 + d) * 2048 + t] = (__bf16)val;
          vout[((long long)((bb * 16 + h) * 2048 + t)) * 128 + d] = val;
        }
      }
    }
  }
}

// Flash-style causal attention. One wave per (b, h, 16-row q tile).
__global__ void __launch_bounds__(256) attn_kernel(
    const __bf16* __restrict__ qf, const __bf16* __restrict__ kf,
    const __bf16* __restrict__ vT, __bf16* __restrict__ ybf) {
  const int lane = threadIdx.x & 31;
  const int wv = threadIdx.x >> 5;
  const int wid = (blockIdx.x * blockDim.x + threadIdx.x) >> 5;  // 0..8191
  const int QT = 2048 / 16;
  const int qt = wid % QT;
  const int bh = wid / QT;  // b*16 + h
  const int q0 = qt * 16;
  const int h = bh & 15, b = bh >> 4;
  const int colh = h * 128;
  const int rowb = b * 2048;

  __shared__ __align__(16) __bf16 pbuf[8][16 * 32];  // per-wave 16x32 P tile

  v16bf qfr[4];
#pragma unroll
  for (int i = 0; i < 4; ++i)
    qfr[i] = load_frag(qf, rowb + q0, 2048, colh + 32 * i, lane);

  v8f acc[8] = {};
  float mrun[8], lrun[8];
#pragma unroll
  for (int j = 0; j < 8; ++j) { mrun[j] = NEG_BIG; lrun[j] = 0.f; }

  const int half = lane >> 4, n = lane & 15;
  const int ntiles = (q0 + 15) / 32 + 1;  // causal: skip future kv tiles

  for (int kt = 0; kt < ntiles; ++kt) {
    const int kv0 = kt * 32;
    v8f s[2] = {};
#pragma unroll
    for (int c = 0; c < 2; ++c) {
#pragma unroll
      for (int i = 0; i < 4; ++i) {
        v16bf bfr = load_frag(kf, rowb + kv0 + 16 * c, 2048, colh + 32 * i, lane);
        s[c] = wmma_bf16(qfr[i], bfr, s[c]);
      }
    }
    const bool needmask = (kv0 + 31) > q0;
#pragma unroll
    for (int j = 0; j < 8; ++j) {
      const int m = j + 8 * half;
      float s0 = s[0][j], s1 = s[1][j];
      if (needmask) {
        if (kv0 + n > q0 + m) s0 = NEG_BIG;
        if (kv0 + 16 + n > q0 + m) s1 = NEG_BIG;
      }
      float tmax = fmaxf(s0, s1);
      tmax = fmaxf(tmax, __shfl_xor(tmax, 1, 16));
      tmax = fmaxf(tmax, __shfl_xor(tmax, 2, 16));
      tmax = fmaxf(tmax, __shfl_xor(tmax, 4, 16));
      tmax = fmaxf(tmax, __shfl_xor(tmax, 8, 16));
      const float newm = fmaxf(mrun[j], tmax);
      const float alpha = __expf(mrun[j] - newm);
      const float p0 = __expf(s0 - newm);
      const float p1 = __expf(s1 - newm);
      float ps = p0 + p1;
      ps += __shfl_xor(ps, 1, 16);
      ps += __shfl_xor(ps, 2, 16);
      ps += __shfl_xor(ps, 4, 16);
      ps += __shfl_xor(ps, 8, 16);
      lrun[j] = lrun[j] * alpha + ps;
      mrun[j] = newm;
#pragma unroll
      for (int t = 0; t < 8; ++t) acc[t][j] *= alpha;
      pbuf[wv][m * 32 + n] = (__bf16)p0;
      pbuf[wv][m * 32 + 16 + n] = (__bf16)p1;
    }
    asm volatile("s_wait_dscnt 0" ::: "memory");  // wave-local LDS visibility
    {
      const int r = lane & 15;
      const __bf16* pp = &pbuf[wv][r * 32 + half * 8];
      v8bf lo = *(const v8bf*)pp;
      v8bf hi = *(const v8bf*)(pp + 16);
      v16bf pf = __builtin_shufflevector(lo, hi, 0, 1, 2, 3, 4, 5, 6, 7,
                                         8, 9, 10, 11, 12, 13, 14, 15);
      asm volatile("s_wait_dscnt 0" ::: "memory");
#pragma unroll
      for (int t = 0; t < 8; ++t) {
        v16bf vfr = load_frag(vT, bh * 128 + t * 16, 2048, kv0, lane);
        acc[t] = wmma_bf16(pf, vfr, acc[t]);
      }
    }
  }
#pragma unroll
  for (int j = 0; j < 8; ++j) {
    const int m = j + 8 * half;
    const float inv = 1.0f / lrun[j];
    const long long row = (long long)(rowb + q0 + m);
#pragma unroll
    for (int t = 0; t < 8; ++t)
      ybf[row * 2048 + colh + t * 16 + n] = (__bf16)(acc[t][j] * inv);
  }
}

// Output projection: M=8192, N=2048, K=2048, fp32 epilogue to d_out.
__global__ void __launch_bounds__(256) proj_kernel(
    const __bf16* __restrict__ ybf, const __bf16* __restrict__ wbf,
    const float* __restrict__ bias, float* __restrict__ yout) {
  __shared__ __align__(16) __bf16 shA[DEPTH * 128 * LDSROW];
  __shared__ __align__(16) __bf16 shB[DEPTH * 128 * LDSROW];
  const int tm = blockIdx.x / 16, tn = blockIdx.x % 16;
  const int row0 = tm * 128, col0 = tn * 128;
  v8f acc[2][4] = {};
  gemm_mainloop(ybf, wbf, row0, col0, shA, shB, acc);

  const int lane = threadIdx.x & 31;
  const int wv = threadIdx.x >> 5;
  const int wm = (wv >> 1) * 32, wn = (wv & 1) * 64;
  const int n = lane & 15, half = lane >> 4;
#pragma unroll
  for (int i = 0; i < 2; ++i) {
#pragma unroll
    for (int j = 0; j < 4; ++j) {
#pragma unroll
      for (int jj = 0; jj < 8; ++jj) {
        const int o = col0 + wn + 16 * j + n;
        const long long row = row0 + wm + 16 * i + jj + 8 * half;
        yout[row * 2048 + o] = acc[i][j][jj] + bias[o];
      }
    }
  }
}

extern "C" void kernel_launch(void* const* d_in, const int* in_sizes, int n_in,
                              void* d_out, int out_size, void* d_ws,
                              size_t ws_size, hipStream_t stream) {
  const float* x      = (const float*)d_in[0];
  const float* w_attn = (const float*)d_in[1];
  const float* b_attn = (const float*)d_in[2];
  const float* w_proj = (const float*)d_in[3];
  const float* b_proj = (const float*)d_in[4];

  float* out = (float*)d_out;
  const long long Ysz = 4LL * 2048 * 2048;
  float* kout = out + Ysz;
  float* vout = out + 2 * Ysz;

  char* ws = (char*)d_ws;
  __bf16* xbf  = (__bf16*)(ws);              // 33554432 B
  __bf16* wabf = (__bf16*)(ws + 33554432);   // 25165824 B
  __bf16* wpbf = (__bf16*)(ws + 58720256);   //  8388608 B
  __bf16* qbf  = (__bf16*)(ws + 67108864);   // 33554432 B
  __bf16* kbf  = (__bf16*)(ws + 100663296);  // 33554432 B
  __bf16* vTbf = (__bf16*)(ws + 134217728);  // 33554432 B (V^T [b,h,dk,T])
  __bf16* ybf  = (__bf16*)(ws + 167772160);  // 33554432 B

  cvt_kernel<<<4096, 256, 0, stream>>>(x, xbf, 16777216);
  cvt_kernel<<<4096, 256, 0, stream>>>(w_attn, wabf, 12582912);
  cvt_kernel<<<4096, 256, 0, stream>>>(w_proj, wpbf, 4194304);

  // 64 M-blocks * 48 N-blocks, each 128x128
  qkv_kernel<<<3072, 256, 0, stream>>>(xbf, wabf, b_attn, qbf, kbf, vTbf,
                                       kout, vout);
  // 4*16*128 = 8192 waves / 8 per block
  attn_kernel<<<1024, 256, 0, stream>>>(qbf, kbf, vTbf, ybf);
  // 64 * 16 blocks
  proj_kernel<<<1024, 256, 0, stream>>>(ybf, wpbf, b_proj, out);
}
